// MixtureOfExperts_498216206779
// MI455X (gfx1250) — matrix operand order repaired and checked by reading, not compiled, not run on hardware
//
#include <hip/hip_runtime.h>

typedef unsigned short u16;
typedef unsigned int   u32;
typedef unsigned long long u64;

typedef __attribute__((ext_vector_type(16))) __bf16 v16bf;
typedef __attribute__((ext_vector_type(8)))  float  v8f;
typedef __attribute__((ext_vector_type(4)))  unsigned int u32x4;
typedef __attribute__((ext_vector_type(8)))  int i32x8;
typedef __attribute__((ext_vector_type(4)))  int i32x4;

// ---------------- problem constants ----------------
#define NB   4
#define NT   2048
#define ND   1024
#define NH   4096
#define NE   8
#define NN   (NB*NT)      // 8192 tokens
#define CAP  1024         // capacity = ceil(1.0 * N / E)

// ---------------- helpers ----------------
__device__ __forceinline__ u16 f2bf(float f) {
    u32 u = __builtin_bit_cast(u32, f);
    u32 r = u + 0x7FFFu + ((u >> 16) & 1u);   // round-to-nearest-even
    return (u16)(r >> 16);
}

struct Frag32 { uint4 lo; uint4 hi; };

__device__ __forceinline__ v16bf ld_frag(const u16* p0, const u16* p1) {
    Frag32 f;
    f.lo = *reinterpret_cast<const uint4*>(p0);
    f.hi = *reinterpret_cast<const uint4*>(p1);
    return __builtin_bit_cast(v16bf, f);
}

// LDS byte offset of a __shared__ object (flat LDS aperture: low 32 bits)
__device__ __forceinline__ u32 lds_off(const void* p) {
    return (u32)(size_t)p;
}

// ---- Tensor Data Mover: 2D bf16 tile (tile_d0 elems/row x tile_d1 rows),
// global row stride = rstride elems, deposited in LDS with rows padded from
// 16 DWORDs to 24 DWORDs (pad_interval=16DW -> code 3, pad_amount=8DW -> 7),
// i.e. LDS row stride 48 bf16 = 96B. D# layout per CDNA5 ISA ch.8.
__device__ __forceinline__ void tdm_load_2d(u32 ldsAddr, const u16* gptr,
                                            int tile_d0, int tile_d1,
                                            int rstride)
{
    u64 ga = (u64)(size_t)gptr;
    u32x4 g0;
    g0[0] = 1u;                                        // count=1 (valid user D#)
    g0[1] = ldsAddr;                                   // lds_addr (bytes)
    g0[2] = (u32)ga;                                   // global_addr[31:0]
    g0[3] = (u32)((ga >> 32) & 0x1FFFFFFu) | (2u << 30); // ga[56:32] | type=2
    i32x8 g1;
    g1[0] = (1 << 16)                                  // data_size = 2 bytes
          | (1 << 20)                                  // pad_enable
          | (3 << 22)                                  // pad_interval: 16 DW
          | (7 << 25);                                 // pad_amount:    8 DW
    g1[1] = (int)((u32)rstride << 16);                 // tensor_dim0[15:0]<<16
    g1[2] = (int)(((u32)rstride >> 16) & 0xFFFFu) |    // tensor_dim0[31:16]
            (int)((u32)tile_d1 << 16);                 // tensor_dim1[15:0]
    g1[3] = (int)((u32)tile_d0 << 16);                 // td1[31:16]=0 | tile_dim0
    g1[4] = (int)(u32)tile_d1;                         // tile_dim1 | tile_dim2=0
    g1[5] = (int)(u32)rstride;                         // tensor_dim0_stride[31:0]
    g1[6] = 0;                                         // stride[47:32] | td1s lo
    g1[7] = 0;                                         // tensor_dim1_stride hi
    i32x4 g2;
    g2[0] = 1; g2[1] = 1; g2[2] = 0; g2[3] = 0;        // tensor_dim2/3=1, unused
    i32x4 g3 = {0, 0, 0, 0};
    i32x8 g4 = {0, 0, 0, 0, 0, 0, 0, 0};               // unused trailing group
    __builtin_amdgcn_tensor_load_to_lds(g0, g1, g2, g3, g4, 0 /*cpol*/);
}

// ---------------- fill ----------------
__global__ void moe_fill_u32(u32* p, u32 v, long n) {
    long i = (long)blockIdx.x * blockDim.x + threadIdx.x;
    long s = (long)gridDim.x * blockDim.x;
    for (; i < n; i += s) p[i] = v;
}

// ------- weight prep: W [E][K][N] fp32 -> Wt [E][N][K] bf16 (transpose) -----
__global__ void __launch_bounds__(256)
moe_wconv(const float* __restrict__ W, u16* __restrict__ Wt, int K, int N)
{
    __shared__ float t[64][65];
    const int e = blockIdx.z;
    const int kBase = blockIdx.y * 64;
    const int nBase = blockIdx.x * 64;
    const float* Wp = W + (size_t)e * K * N;
    u16* Wtp = Wt + (size_t)e * N * K;
    const int tr = threadIdx.x >> 4;      // 0..15
    const int tc = threadIdx.x & 15;      // 0..15
    #pragma unroll
    for (int i = 0; i < 4; i++) {
        int row = tr + i * 16;            // k
        float4 v = *reinterpret_cast<const float4*>(
            Wp + (size_t)(kBase + row) * N + nBase + tc * 4);
        t[row][tc * 4 + 0] = v.x; t[row][tc * 4 + 1] = v.y;
        t[row][tc * 4 + 2] = v.z; t[row][tc * 4 + 3] = v.w;
    }
    __syncthreads();
    #pragma unroll
    for (int i = 0; i < 4; i++) {
        int nrow = tr + i * 16;           // n
        int kcol = tc * 4;                // k
        uint2 o;
        o.x = (u32)f2bf(t[kcol + 0][nrow]) | ((u32)f2bf(t[kcol + 1][nrow]) << 16);
        o.y = (u32)f2bf(t[kcol + 2][nrow]) | ((u32)f2bf(t[kcol + 3][nrow]) << 16);
        *reinterpret_cast<uint2*>(Wtp + (size_t)(nBase + nrow) * K + kBase + kcol) = o;
    }
}

// ---------------- router: one wave per token ----------------
__global__ void __launch_bounds__(256)
moe_router(const float* __restrict__ x, const float* __restrict__ Wg,
           int* __restrict__ eidx, float* __restrict__ eprob,
           int* __restrict__ cnti, float* __restrict__ blk_psum,
           float* __restrict__ blk_z)
{
    __shared__ float wg[ND * NE];      // 32 KB
    __shared__ float shp[8][8];
    __shared__ float shz[8];
    const int tid = threadIdx.x;
    #pragma unroll
    for (int i = 0; i < 32; i++) wg[tid + 256 * i] = Wg[tid + 256 * i];
    __syncthreads();

    const int wave = tid >> 5, lane = tid & 31;
    const int n = blockIdx.x * 8 + wave;

    float a[NE];
    #pragma unroll
    for (int e = 0; e < NE; e++) a[e] = 0.f;
    const float* xr = x + (size_t)n * ND;
    for (int d = lane; d < ND; d += 32) {
        float xv = xr[d];
        const float* w = wg + d * NE;
        #pragma unroll
        for (int e = 0; e < NE; e++) a[e] = fmaf(xv, w[e], a[e]);
    }
    #pragma unroll
    for (int off = 16; off >= 1; off >>= 1)
        #pragma unroll
        for (int e = 0; e < NE; e++) a[e] += __shfl_xor(a[e], off, 32);

    if (lane == 0) {
        float m = a[0]; int idx = 0;
        #pragma unroll
        for (int e = 1; e < NE; e++) if (a[e] > m) { m = a[e]; idx = e; }
        float p[NE], s = 0.f;
        #pragma unroll
        for (int e = 0; e < NE; e++) { p[e] = __expf(a[e] - m); s += p[e]; }
        float inv = 1.f / s;
        float lse = m + __logf(s);
        eidx[n]  = idx;
        eprob[n] = p[idx] * inv;
        atomicAdd(&cnti[idx], 1);                 // integer: deterministic
        #pragma unroll
        for (int e = 0; e < NE; e++) shp[wave][e] = p[e] * inv;
        shz[wave] = lse * lse;
    }
    __syncthreads();
    if (tid < 8) {                                 // deterministic partials
        float s = 0.f;
        #pragma unroll
        for (int w = 0; w < 8; w++) s += shp[w][tid];
        blk_psum[blockIdx.x * 8 + tid] = s;
    }
    if (tid == 8) {
        float s = 0.f;
        #pragma unroll
        for (int w = 0; w < 8; w++) s += shz[w];
        blk_z[blockIdx.x] = s;
    }
}

__global__ void moe_reduce(const float* __restrict__ blk_psum,
                           const float* __restrict__ blk_z,
                           float* __restrict__ psum, float* __restrict__ zsum)
{
    int t = threadIdx.x;
    if (t < 8) {
        float s = 0.f;
        for (int b = 0; b < NN / 8; b++) s += blk_psum[b * 8 + t];
        psum[t] = s;
    } else if (t == 8) {
        float s = 0.f;
        for (int b = 0; b < NN / 8; b++) s += blk_z[b];
        zsum[0] = s;
    }
}

// ---------------- stable per-expert ranking (single block) ----------------
__global__ void __launch_bounds__(256)
moe_dispatch(const int* __restrict__ eidx, const float* __restrict__ eprob,
             const int* __restrict__ cnti, int* __restrict__ rank,
             int* __restrict__ tok, float* __restrict__ scale)
{
    __shared__ int le[256];
    __shared__ int base[NE];
    __shared__ int startS[NE];
    const int t = threadIdx.x;
    if (t < NE) base[t] = 0;
    if (t == 0) {
        int s = 0;
        for (int e = 0; e < NE; e++) { startS[e] = s; s += cnti[e]; }
    }
    __syncthreads();
    for (int ch = 0; ch < NN / 256; ch++) {
        int n = ch * 256 + t;
        int e = eidx[n];
        le[t] = e;
        __syncthreads();
        int rl = 0;
        for (int j = 0; j < t; j++) rl += (le[j] == e);   // stable local rank
        int r = base[e] + rl;
        rank[n] = r;
        if (r < CAP) tok[e * CAP + r] = n;                // slot -> token
        scale[startS[e] + r] = eprob[n];                  // quirky sorted prob
        __syncthreads();
        if (t < NE) {
            int c = 0;
            for (int j = 0; j < 256; j++) c += (le[j] == t);
            base[t] += c;
        }
        __syncthreads();
    }
}

// ---------------- scatter tokens into bf16 capacity buffer ----------------
__global__ void __launch_bounds__(256)
moe_scatter(const float* __restrict__ x, const int* __restrict__ eidx,
            const int* __restrict__ rank, u16* __restrict__ bufbf)
{
    const int n = blockIdx.x;
    const int r = rank[n];
    if (r >= CAP) return;                       // dropped by capacity
    const int e = eidx[n];
    float4 v = reinterpret_cast<const float4*>(x + (size_t)n * ND)[threadIdx.x];
    uint2 o;
    o.x = (u32)f2bf(v.x) | ((u32)f2bf(v.y) << 16);
    o.y = (u32)f2bf(v.z) | ((u32)f2bf(v.w) << 16);
    reinterpret_cast<uint2*>(bufbf + ((size_t)e * CAP + r) * ND)[threadIdx.x] = o;
}

// ---------------- TDM double-buffered WMMA grouped-GEMM core ----------------
// Block: 256 threads (8 waves, 2x4), output tile 128(M) x 256(N), K step 32.
// Both A (bf16 [M][K]) and B (bf16 [N][K], pre-transposed) tiles are moved by
// the Tensor Data Mover into LDS with rows padded to 48 bf16 (96B).
#define LDT 48
#define SA_ELEMS (128 * LDT)
#define SB_ELEMS (256 * LDT)

template <int KDIM>
__device__ __forceinline__ void
gemm_core(const u16* __restrict__ A, const u16* __restrict__ Bt,
          int mBase, int nBase, u16* sA, u16* sB, v8f acc[16])
{
    const int tid  = threadIdx.x;
    const int lane = tid & 31;
    const int wave = tid >> 5;
    const int wm = (wave >> 2) * 64;
    const int wn = (wave & 3) * 64;
    const int lm = lane & 15;
    const int lh = lane >> 4;

    #pragma unroll
    for (int i = 0; i < 16; i++) acc[i] = v8f{};

    constexpr int STEPS = KDIM / 32;
    const u16* Arow = A + (size_t)mBase * KDIM;
    const u16* Brow = Bt + (size_t)nBase * KDIM;

    if (wave == 0) {                               // stage 0 prefetch
        tdm_load_2d(lds_off(sA), Arow, 32, 128, KDIM);
        tdm_load_2d(lds_off(sB), Brow, 32, 256, KDIM);
    }
    for (int s = 0; s < STEPS; s++) {
        if (wave == 0) {
            if (s + 1 < STEPS) {                   // prefetch next stage
                int kk = (s + 1) * 32;
                u16* nA = sA + ((s + 1) & 1) * SA_ELEMS;
                u16* nB = sB + ((s + 1) & 1) * SB_ELEMS;
                tdm_load_2d(lds_off(nA), Arow + kk, 32, 128, KDIM);
                tdm_load_2d(lds_off(nB), Brow + kk, 32, 256, KDIM);
                __builtin_amdgcn_s_wait_tensorcnt(2);  // stage s resident
            } else {
                __builtin_amdgcn_s_wait_tensorcnt(0);
            }
        }
        __syncthreads();                            // publish TDM-filled LDS

        const u16* cA = sA + (s & 1) * SA_ELEMS;
        const u16* cB = sB + (s & 1) * SB_ELEMS;
        v16bf af[4];
        #pragma unroll
        for (int mi = 0; mi < 4; mi++) {
            int r = wm + mi * 16 + lm;
            af[mi] = ld_frag(cA + r * LDT + lh * 8,
                             cA + r * LDT + 16 + lh * 8);
        }
        #pragma unroll
        for (int ni = 0; ni < 4; ni++) {
            int nc = wn + ni * 16 + lm;
            v16bf bf = ld_frag(cB + nc * LDT + lh * 8,
                               cB + nc * LDT + 16 + lh * 8);
            #pragma unroll
            for (int mi = 0; mi < 4; mi++) {
                acc[mi * 4 + ni] = __builtin_amdgcn_wmma_f32_16x16x32_bf16(
                    false, af[mi], false, bf, (short)0, acc[mi * 4 + ni],
                    false, false);
            }
        }
        __syncthreads();                            // done reading buffer s&1
    }
}

// ---------------- GEMM1: h = gelu(buf @ W1 + b1), bf16 out ----------------
__global__ void __launch_bounds__(256)
moe_gemm1(const u16* __restrict__ bufbf, const u16* __restrict__ Wt1,
          const float* __restrict__ b1, u16* __restrict__ hbf)
{
    __shared__ __align__(128) u16 sA[2 * SA_ELEMS];
    __shared__ __align__(128) u16 sB[2 * SB_ELEMS];
    const int e = blockIdx.z;
    const int mBase = blockIdx.y * 128;
    const int nBase = blockIdx.x * 256;

    v8f acc[16];
    gemm_core<ND>(bufbf + (size_t)e * CAP * ND,
                  Wt1 + (size_t)e * NH * ND, mBase, nBase, sA, sB, acc);

    const int lane = threadIdx.x & 31;
    const int wave = threadIdx.x >> 5;
    const int wm = (wave >> 2) * 64, wn = (wave & 3) * 64;
    const int lm = lane & 15, lh = lane >> 4;

    float bias[4];
    #pragma unroll
    for (int ni = 0; ni < 4; ni++)
        bias[ni] = b1[(size_t)e * NH + nBase + wn + ni * 16 + lm];

    #pragma unroll
    for (int mi = 0; mi < 4; mi++)
        #pragma unroll
        for (int ni = 0; ni < 4; ni++)
            #pragma unroll
            for (int r = 0; r < 8; r++) {
                float v = acc[mi * 4 + ni][r] + bias[ni];
                v = 0.5f * v * (1.0f + erff(v * 0.70710678118654752f)); // exact gelu
                int row = mBase + wm + mi * 16 + r + 8 * lh;
                int col = nBase + wn + ni * 16 + lm;
                hbf[(size_t)e * CAP * NH + (size_t)row * NH + col] = f2bf(v);
            }
}

// ------- GEMM2: o = h @ W2 + b2, fused combine (unsort + quirky scale) ------
__global__ void __launch_bounds__(256)
moe_gemm2(const u16* __restrict__ hbf, const u16* __restrict__ Wt2,
          const float* __restrict__ b2, const int* __restrict__ tok,
          const float* __restrict__ scale, const int* __restrict__ cnti,
          float* __restrict__ out)
{
    __shared__ __align__(128) u16 sA[2 * SA_ELEMS];
    __shared__ __align__(128) u16 sB[2 * SB_ELEMS];
    __shared__ int   tokS[128];
    __shared__ float sclS[128];
    const int e = blockIdx.z;
    const int mBase = blockIdx.y * 128;
    const int nBase = blockIdx.x * 256;

    if (threadIdx.x < 128) {
        int slot = mBase + threadIdx.x;
        int ce = cnti[e];
        int t = (slot < ce) ? tok[e * CAP + slot] : -1;
        tokS[threadIdx.x] = t;
        sclS[threadIdx.x] = (t >= 0) ? scale[t] : 0.f;
    }
    v8f acc[16];
    gemm_core<NH>(hbf + (size_t)e * CAP * NH,
                  Wt2 + (size_t)e * ND * NH, mBase, nBase, sA, sB, acc);

    const int lane = threadIdx.x & 31;
    const int wave = threadIdx.x >> 5;
    const int wm = (wave >> 2) * 64, wn = (wave & 3) * 64;
    const int lm = lane & 15, lh = lane >> 4;

    float bias[4];
    #pragma unroll
    for (int ni = 0; ni < 4; ni++)
        bias[ni] = b2[(size_t)e * ND + nBase + wn + ni * 16 + lm];

    #pragma unroll
    for (int mi = 0; mi < 4; mi++)
        #pragma unroll
        for (int r = 0; r < 8; r++) {
            int rl = wm + mi * 16 + r + 8 * lh;
            int t = tokS[rl];
            if (t >= 0) {
                float sc = sclS[rl];
                #pragma unroll
                for (int ni = 0; ni < 4; ni++) {
                    float v = acc[mi * 4 + ni][r] + bias[ni];
                    int col = nBase + wn + ni * 16 + lm;
                    out[(size_t)t * ND + col] = v * sc;
                }
            }
        }
}

// ---------------- aux loss scalar ----------------
__global__ void moe_aux(const int* __restrict__ cnti,
                        const float* __restrict__ psum,
                        const float* __restrict__ zsum,
                        float* __restrict__ dst)
{
    if (threadIdx.x == 0 && blockIdx.x == 0) {
        float s = 0.f;
        for (int e = 0; e < NE; e++)
            s += ((float)cnti[e] / (float)NN) * (psum[e] / (float)NN);
        dst[0] = 0.01f * (float)NE * s + 0.001f * (zsum[0] / (float)NN);
    }
}

// ---------------- host side ----------------
extern "C" void kernel_launch(void* const* d_in, const int* in_sizes, int n_in,
                              void* d_out, int out_size, void* d_ws, size_t ws_size,
                              hipStream_t stream) {
    const float* x  = (const float*)d_in[0];
    const float* Wg = (const float*)d_in[1];
    const float* W1 = (const float*)d_in[2];
    const float* b1 = (const float*)d_in[3];
    const float* W2 = (const float*)d_in[4];
    const float* b2 = (const float*)d_in[5];
    float* out = (float*)d_out;

    char* ws = (char*)d_ws;
    int*   cnti = (int*)(ws + 0);                       // 8 ints
    float* psum = (float*)(ws + 32);                    // 8 floats
    float* zsum = (float*)(ws + 64);                    // 1 float
    size_t off = 256;
    int*   eidx  = (int*)  (ws + off); off += (size_t)NN * 4;
    float* eprob = (float*)(ws + off); off += (size_t)NN * 4;
    int*   rank  = (int*)  (ws + off); off += (size_t)NN * 4;
    float* scale = (float*)(ws + off); off += (size_t)NN * 4;
    int*   tok   = (int*)  (ws + off); off += (size_t)NE * CAP * 4;
    float* blkp  = (float*)(ws + off); off += (size_t)NN * 4;
    float* blkz  = (float*)(ws + off); off += (size_t)(NN / 8) * 4;
    u16*   bufbf = (u16*)  (ws + off); off += (size_t)NE * CAP * ND * 2; // 16MB
    u16*   hbf   = (u16*)  (ws + off); off += (size_t)NE * CAP * NH * 2; // 64MB
    u16*   Wt1   = (u16*)  (ws + off); off += (size_t)NE * ND * NH * 2;  // 64MB
    u16*   Wt2   = (u16*)  (ws + off); off += (size_t)NE * NH * ND * 2;  // 64MB

    // 1) zero output, dispatch buffer, counters
    moe_fill_u32<<<2048, 256, 0, stream>>>((u32*)out, 0u, (long)NN * ND + 1);
    moe_fill_u32<<<2048, 256, 0, stream>>>((u32*)bufbf, 0u, (long)NE * CAP * ND / 2);
    moe_fill_u32<<<1, 32, 0, stream>>>((u32*)cnti, 0u, 8);

    // 2) weight prep: fp32 -> bf16 + transpose (K-contiguous for TDM)
    moe_wconv<<<dim3(NH / 64, ND / 64, NE), 256, 0, stream>>>(W1, Wt1, ND, NH);
    moe_wconv<<<dim3(ND / 64, NH / 64, NE), 256, 0, stream>>>(W2, Wt2, NH, ND);

    // 3) router
    moe_router<<<NN / 8, 256, 0, stream>>>(x, Wg, eidx, eprob, cnti, blkp, blkz);
    moe_reduce<<<1, 32, 0, stream>>>(blkp, blkz, psum, zsum);

    // 4) stable ranking / dispatch metadata
    moe_dispatch<<<1, 256, 0, stream>>>(eidx, eprob, cnti, rank, tok, scale);

    // 5) scatter tokens -> bf16 capacity buffer
    moe_scatter<<<NN, 256, 0, stream>>>(x, eidx, rank, bufbf);

    // 6) expert GEMM 1 (+gelu)  [E,1024,1024] x [E,1024,4096]
    moe_gemm1<<<dim3(NH / 256, CAP / 128, NE), 256, 0, stream>>>(bufbf, Wt1, b1, hbf);

    // 7) expert GEMM 2 + fused combine  [E,1024,4096] x [E,4096,1024]
    moe_gemm2<<<dim3(ND / 256, CAP / 128, NE), 256, 0, stream>>>(hbf, Wt2, b2, tok,
                                                                 scale, cnti, out);

    // 8) aux loss scalar -> out[N*D]
    moe_aux<<<1, 32, 0, stream>>>(cnti, psum, zsum, out + (size_t)NN * ND);
}
